// TNTGlobalBranch_35064113004736
// MI455X (gfx1250) — matrix operand order repaired and checked by reading, not compiled, not run on hardware
//
#include <hip/hip_runtime.h>

// LFLTM fast-weight scan for MI455X (gfx1250, wave32).
// Row-decomposed: one wave owns one (batch, row) state scan; 4 waves per
// workgroup share the LDS-staged X chunk. The intra-chunk cumsum is
// reformulated as a per-row kernel matrix K_i = (x .* lr_i) @ X^T computed
// with v_wmma_f32_16x16x32_bf16 (lower-triangular tiles only), followed by a
// small masked contraction with err. X chunks are staged Global->LDS with the
// Tensor Data Mover (pad feature produces conflict-padded LDS rows).
// B operands come from a token-major bf16 copy (contiguous b128 LDS loads,
// no transpose buffer needed); A tiles (x*lr) are built once per s-tile and
// held in VGPRs across the t/k loops.

#define DIMD    128
#define CHUNKC  64
#define NCHNK   128
#define SEQL    8192
#define BASE_LR 0.005f
#define META_LR 0.01f

#define XF_LD 132   // padded f32 row stride (128 + 4) -> matches TDM pad
#define XH_LD 136   // padded bf16 row stride (128 + 8)

typedef __attribute__((ext_vector_type(16))) __bf16       v16bf;
typedef __attribute__((ext_vector_type(8)))  float        v8f;
typedef __attribute__((ext_vector_type(4)))  unsigned int v4u;
typedef __attribute__((ext_vector_type(8)))  int          v8i;
typedef __attribute__((ext_vector_type(4)))  int          v4i;

#if defined(__has_builtin)
#  if __has_builtin(__builtin_amdgcn_tensor_load_to_lds) && \
      __has_builtin(__builtin_amdgcn_s_wait_tensorcnt)
#    define USE_TDM 1
#  endif
#endif
#ifndef USE_TDM
#  define USE_TDM 0
#endif

__global__ __launch_bounds__(128, 1) void
lfltm_fastweight_scan(const float* __restrict__ X,
                      const float* __restrict__ init_mem,
                      const float* __restrict__ init_opt,
                      float* __restrict__ out)
{
  __shared__ __align__(16) float  Xf[CHUNKC * XF_LD];  // f32 chunk, padded rows
  __shared__ __align__(16) __bf16 Xh[CHUNKC * XH_LD];  // bf16 copy, token-major
  __shared__ __align__(16) float  Wrow[4][DIMD];       // per-wave fast-weight row
  __shared__ __align__(16) float  Orow[4][DIMD];       // per-wave lr-logit row
  __shared__ __align__(16) float  Lrow[4][DIMD];       // sigmoid(opt), chunk start
  __shared__ __align__(16) float  Erow[4][CHUNKC];     // per-token err, this row

  const int tid  = threadIdx.x;
  const int wave = tid >> 5;
  const int lane = tid & 31;
  const int half = lane >> 4;
  const int nn   = lane & 15;
  const int b    = blockIdx.y;
  const int i    = blockIdx.x * 4 + wave;   // row of W owned by this wave

  for (int j = lane; j < DIMD; j += 32) {
    Wrow[wave][j] = init_mem[i * DIMD + j];
    Orow[wave][j] = init_opt[i * DIMD + j];
  }

  const float* Xb = X   + (size_t)b * SEQL * DIMD;
  float*       Ob = out + (size_t)b * SEQL * DIMD;

  for (int c = 0; c < NCHNK; ++c) {
    __syncthreads();  // previous chunk's Xf/Xh fully consumed
    const float* src = Xb + (size_t)c * CHUNKC * DIMD;

#if USE_TDM
    if (wave == 0) {
      unsigned long long ga = (unsigned long long)(const void*)src;
      unsigned int lo = (unsigned int)(unsigned long long)(void*)&Xf[0];
      v4u g0;
      g0[0] = 1u;                                   // count = 1 (valid D#)
      g0[1] = lo;                                   // lds_addr
      g0[2] = (unsigned int)ga;                     // global_addr[31:0]
      g0[3] = (unsigned int)((ga >> 32) & 0x01FFFFFFull) | (2u << 30); // +type=2
      v8i g1;
      // data_size=4B, pad_enable, pad_interval=128 DW, pad_amount=4 DW
      g1[0] = (int)((2u << 16) | (1u << 20) | (6u << 22) | (3u << 25));
      g1[1] = (int)(128u << 16);                    // tensor_dim0 = 128
      g1[2] = (int)(64u  << 16);                    // tensor_dim1 = 64
      g1[3] = (int)(128u << 16);                    // tile_dim0 = 128
      g1[4] = 64;                                   // tile_dim1 = 64
      g1[5] = 128;                                  // tensor_dim0_stride = 128
      g1[6] = 0;
      g1[7] = 0;
      v4i z4 = {};
      v8i z8 = {};
      __builtin_amdgcn_tensor_load_to_lds(g0, g1, z4, z4, z8, 0);
      __builtin_amdgcn_s_wait_tensorcnt((short)0);
    }
#else
    for (int idx = tid * 4; idx < CHUNKC * DIMD; idx += 128 * 4) {
      int s = idx >> 7, j = idx & (DIMD - 1);
      const float4 vv = *(const float4*)(src + idx);
      *(float4*)&Xf[s * XF_LD + j] = vv;
    }
#endif
    __syncthreads();

    // token-major bf16 copy (4 elements per thread per step, b64 stores)
    for (int idx = tid * 4; idx < CHUNKC * DIMD; idx += 128 * 4) {
      int s = idx >> 7, j = idx & (DIMD - 1);
      float4 xv = *(const float4*)&Xf[s * XF_LD + j];
      __bf16* dst = &Xh[s * XH_LD + j];
      dst[0] = (__bf16)xv.x; dst[1] = (__bf16)xv.y;
      dst[2] = (__bf16)xv.z; dst[3] = (__bf16)xv.w;
    }
    __syncthreads();

    // chunk-start learning rates
    for (int j = lane; j < DIMD; j += 32)
      Lrow[wave][j] = 1.0f / (1.0f + __expf(-Orow[wave][j]));

    // err_t = W_i . x_t - x_{t,i}  (two tokens per lane, float4 loads)
    {
      const int t0 = lane, t1 = 32 + lane;
      float e0 = 0.f, e1 = 0.f;
      #pragma unroll 8
      for (int jb = 0; jb < DIMD / 4; ++jb) {
        float4 w  = *(const float4*)&Wrow[wave][jb * 4];
        float4 xa = *(const float4*)&Xf[t0 * XF_LD + jb * 4];
        float4 xc = *(const float4*)&Xf[t1 * XF_LD + jb * 4];
        e0 = fmaf(w.x, xa.x, fmaf(w.y, xa.y, fmaf(w.z, xa.z, fmaf(w.w, xa.w, e0))));
        e1 = fmaf(w.x, xc.x, fmaf(w.y, xc.y, fmaf(w.z, xc.z, fmaf(w.w, xc.w, e1))));
      }
      Erow[wave][t0] = e0 - Xf[t0 * XF_LD + i];
      Erow[wave][t1] = e1 - Xf[t1 * XF_LD + i];
    }
    __syncthreads();  // publish Lrow/Erow across lanes

    // outc[t] = sum_{s<=t} err_s * K_i(s,t), K_i = (X .* lr_i) @ X^T via WMMA.
    // st outer: build the 4 A tiles (one per kt) once, hold in VGPRs.
    float outc[4] = {0.f, 0.f, 0.f, 0.f};
    #pragma unroll
    for (int st = 0; st < 4; ++st) {
      v16bf atile[4];
      const int srow = st * 16 + nn;               // A: lane m = nn
      #pragma unroll
      for (int kt = 0; kt < 4; ++kt) {
        const int kb = kt * 32 + half * 8;         // halves split K
        float4 x0 = *(const float4*)&Xf[srow * XF_LD + kb];
        float4 x1 = *(const float4*)&Xf[srow * XF_LD + kb + 4];
        float4 x2 = *(const float4*)&Xf[srow * XF_LD + kb + 16];
        float4 x3 = *(const float4*)&Xf[srow * XF_LD + kb + 20];
        float4 l0 = *(const float4*)&Lrow[wave][kb];
        float4 l1 = *(const float4*)&Lrow[wave][kb + 4];
        float4 l2 = *(const float4*)&Lrow[wave][kb + 16];
        float4 l3 = *(const float4*)&Lrow[wave][kb + 20];
        v16bf a;
        a[0]  = (__bf16)(x0.x * l0.x); a[1]  = (__bf16)(x0.y * l0.y);
        a[2]  = (__bf16)(x0.z * l0.z); a[3]  = (__bf16)(x0.w * l0.w);
        a[4]  = (__bf16)(x1.x * l1.x); a[5]  = (__bf16)(x1.y * l1.y);
        a[6]  = (__bf16)(x1.z * l1.z); a[7]  = (__bf16)(x1.w * l1.w);
        a[8]  = (__bf16)(x2.x * l2.x); a[9]  = (__bf16)(x2.y * l2.y);
        a[10] = (__bf16)(x2.z * l2.z); a[11] = (__bf16)(x2.w * l2.w);
        a[12] = (__bf16)(x3.x * l3.x); a[13] = (__bf16)(x3.y * l3.y);
        a[14] = (__bf16)(x3.z * l3.z); a[15] = (__bf16)(x3.w * l3.w);
        atile[kt] = a;
      }
      #pragma unroll
      for (int tt = st; tt < 4; ++tt) {
        v8f acc = {};
        const int tcol = tt * 16 + nn;             // B: lane n = nn
        #pragma unroll
        for (int kt = 0; kt < 4; ++kt) {
          const int kb2 = kt * 32 + half * 8;      // halves split K
          union { uint4 q[2]; v16bf v; } bu;       // 16 bf16 = two 8-wide runs (k, k+16), 2xb128
          bu.q[0] = *(const uint4*)&Xh[tcol * XH_LD + kb2];
          bu.q[1] = *(const uint4*)&Xh[tcol * XH_LD + kb2 + 16];
          acc = __builtin_amdgcn_wmma_f32_16x16x32_bf16(
              false, atile[kt], false, bu.v, (short)0, acc, false, false);
        }
        // masked contraction with err over the M (=s) dimension
        float p = 0.f;
        #pragma unroll
        for (int v = 0; v < 8; ++v) {
          const int sl = half * 8 + v;             // s within tile
          float w = Erow[wave][st * 16 + sl];
          if (st == tt && sl > nn) w = 0.f;        // triangular mask s <= t
          p = fmaf(w, acc[v], p);
        }
        p += __shfl_xor(p, 16);                    // combine M-halves
        outc[tt] += p;
      }
    }

    // out_{t,i} = err_t + x_{t,i} - BASE_LR * outc[t]
    if (half == 0) {
      #pragma unroll
      for (int tt = 0; tt < 4; ++tt) {
        int t = tt * 16 + nn;
        float val = Erow[wave][t] + Xf[t * XF_LD + i] - BASE_LR * outc[tt];
        Ob[(size_t)(c * CHUNKC + t) * DIMD + i] = val;
      }
    }

    // carry: W_i -= BASE_LR*lr.*(err^T X), opt_i -= META_LR*lr(1-lr).*(err2^T X2)
    {
      const int j0 = lane * 4;                     // 4 consecutive j per lane
      float4 S = {0.f, 0.f, 0.f, 0.f};
      float4 T = {0.f, 0.f, 0.f, 0.f};
      #pragma unroll 4
      for (int t = 0; t < CHUNKC; ++t) {
        float  e  = Erow[wave][t];
        float  e2 = e * e;
        float4 x  = *(const float4*)&Xf[t * XF_LD + j0];
        S.x = fmaf(e, x.x, S.x);   S.y = fmaf(e, x.y, S.y);
        S.z = fmaf(e, x.z, S.z);   S.w = fmaf(e, x.w, S.w);
        T.x = fmaf(e2, x.x * x.x, T.x); T.y = fmaf(e2, x.y * x.y, T.y);
        T.z = fmaf(e2, x.z * x.z, T.z); T.w = fmaf(e2, x.w * x.w, T.w);
      }
      float4 lr = *(const float4*)&Lrow[wave][j0];
      float4 w  = *(const float4*)&Wrow[wave][j0];
      float4 o  = *(const float4*)&Orow[wave][j0];
      w.x -= BASE_LR * lr.x * S.x;  w.y -= BASE_LR * lr.y * S.y;
      w.z -= BASE_LR * lr.z * S.z;  w.w -= BASE_LR * lr.w * S.w;
      o.x -= META_LR * lr.x * (1.f - lr.x) * T.x;
      o.y -= META_LR * lr.y * (1.f - lr.y) * T.y;
      o.z -= META_LR * lr.z * (1.f - lr.z) * T.z;
      o.w -= META_LR * lr.w * (1.f - lr.w) * T.w;
      *(float4*)&Wrow[wave][j0] = w;
      *(float4*)&Orow[wave][j0] = o;
    }
  }
}

extern "C" void kernel_launch(void* const* d_in, const int* in_sizes, int n_in,
                              void* d_out, int out_size, void* d_ws, size_t ws_size,
                              hipStream_t stream) {
  (void)in_sizes; (void)n_in; (void)out_size; (void)d_ws; (void)ws_size;
  const float* X        = (const float*)d_in[0];
  const float* init_mem = (const float*)d_in[1];
  const float* init_opt = (const float*)d_in[2];
  float*       out      = (float*)d_out;
  dim3 grid(32, 16);   // 32 row-blocks (4 rows each) x 16 batches = 512 WGs
  dim3 block(128);     // 4 waves, one row per wave
  hipLaunchKernelGGL(lfltm_fastweight_scan, grid, block, 0, stream,
                     X, init_mem, init_opt, out);
}
